// IntraBlockAttention_64355789963666
// MI455X (gfx1250) — compile-verified, matrix operand order checked by reading, not run on hardware
//
#include <hip/hip_runtime.h>
#include <hip/hip_bf16.h>

// ---------------- problem constants ----------------
#define BB   8
#define NN   512
#define HIDD 512
#define HH   8
#define EE   32
#define DD   64
// SCALE = sqrt(64) = 8 -> 1/SCALE = 0.125f

typedef _Float16 h16;
typedef __attribute__((ext_vector_type(16))) _Float16 v16h;
typedef __attribute__((ext_vector_type(8)))  _Float16 v8h;
typedef __attribute__((ext_vector_type(8)))  float    v8f;
typedef __attribute__((ext_vector_type(4)))  unsigned int u32x4;
typedef __attribute__((ext_vector_type(8)))  int      i32x8;
typedef __attribute__((ext_vector_type(4)))  int      i32x4;

union V16H { v16h v; v8h h8[2]; h16 h[16]; };
union V8F  { v8f  v; float f[8]; };

// ---------------- workspace layout (bytes), total ~56 MiB ----------------
static constexpr size_t OFF_XH     = 0;                                  // x f16 row-major: 4 MiB
static constexpr size_t OFF_WQT    = (size_t)4 << 20;                    // W^T f16 (n-major, k contiguous)
static constexpr size_t OFF_WKT    = OFF_WQT + 512 * 512 * 2;
static constexpr size_t OFF_WVT    = OFF_WKT + 512 * 512 * 2;
static constexpr size_t OFF_WOT    = OFF_WVT + 512 * 512 * 2;
static constexpr size_t OFF_WE1T   = (size_t)6 << 20;                    // 64x32 f16 (We1^T)
static constexpr size_t OFF_WE2PT  = OFF_WE1T + 64 * 32 * 2;             // 16x64 f16 (We2^T, rows 8..15 zero)
static constexpr size_t OFF_QH     = (size_t)8 << 20;                    // Q f16 (B,H,N,D): 4 MiB
static constexpr size_t OFF_KH     = (size_t)12 << 20;                   // K f16 (B,H,N,D): 4 MiB
static constexpr size_t OFF_VT     = (size_t)16 << 20;                   // V^T f16 (B,H,D,N): 4 MiB
static constexpr size_t OFF_PRE    = (size_t)20 << 20;                   // context f16 (B*N,512): 4 MiB
static constexpr size_t OFF_EBH    = (size_t)24 << 20;                   // edge bias f16 (B,N,N,H): 32 MiB

// ---------------- WMMA helpers (wave32, 16x16x32 f16) ----------------
// 16-bit A/B fragment for lane l (hi = l>>4): elements are the two contiguous
// 8-half runs k = [8*hi, 8*hi+8) and k = [16+8*hi, 16+8*hi+8)  ->  2 x b128 loads.
__device__ __forceinline__ v16h ldfrag(const h16* p_row_k0, int hi) {
  V16H r;
  r.h8[0] = *(const v8h*)(p_row_k0 + 8 * hi);
  r.h8[1] = *(const v8h*)(p_row_k0 + 16 + 8 * hi);
  return r.v;
}

__device__ __forceinline__ v8f wmma16(v16h a, v16h b, v8f c) {
  return __builtin_amdgcn_wmma_f32_16x16x32_f16(
      false, a, false, b, (short)0, c, false, false);
}

// ---------------- TDM: 2D tile global -> LDS (dsz: 0=1B,1=2B,2=4B) ----------------
__device__ __forceinline__ void tdm_load_2d(unsigned long long gaddr, unsigned int ldsoff,
                                            unsigned int rowElems, unsigned int rows,
                                            unsigned int dsz) {
  u32x4 g0;
  g0[0] = 1u;                                                        // count = 1
  g0[1] = ldsoff;                                                    // lds_addr (bytes)
  g0[2] = (unsigned int)gaddr;                                       // global_addr[31:0]
  g0[3] = (unsigned int)((gaddr >> 32) & 0x1FFFFFFull) | (2u << 30); // addr[56:32] | type=2
  i32x8 g1;
  g1[0] = (int)(dsz << 16);                  // workgroup_mask=0, data_size
  g1[1] = (int)(rowElems << 16);             // tensor_dim0[15:0]  @ bits 63:48
  g1[2] = (int)(rows << 16);                 // tensor_dim1[15:0]  @ bits 95:80
  g1[3] = (int)(rowElems << 16);             // tile_dim0          @ bits 127:112
  g1[4] = (int)rows;                         // tile_dim1          @ bits 143:128
  g1[5] = (int)rowElems;                     // tensor_dim0_stride[31:0]
  g1[6] = 0;
  g1[7] = 0;
  i32x4 z4 = {0, 0, 0, 0};
#if defined(__clang_major__) && (__clang_major__ >= 23)
  i32x8 z8 = {0, 0, 0, 0, 0, 0, 0, 0};
  __builtin_amdgcn_tensor_load_to_lds(g0, g1, z4, z4, z8, 0);
#else
  __builtin_amdgcn_tensor_load_to_lds(g0, g1, z4, z4, 0);
#endif
}

// ---------------- kernel 1: f32 -> f16 staging; weights transposed ----------------
__global__ __launch_bounds__(256) void prep_kernel(const float* __restrict__ x,
                                                   const float* __restrict__ Wq,
                                                   const float* __restrict__ Wk,
                                                   const float* __restrict__ Wv,
                                                   const float* __restrict__ Wo,
                                                   const float* __restrict__ We1,
                                                   const float* __restrict__ We2,
                                                   char* __restrict__ ws) {
  h16* xh  = (h16*)(ws + OFF_XH);
  h16* wt[4] = {(h16*)(ws + OFF_WQT), (h16*)(ws + OFF_WKT),
                (h16*)(ws + OFF_WVT), (h16*)(ws + OFF_WOT)};
  const float* W[4] = {Wq, Wk, Wv, Wo};
  h16* we1t  = (h16*)(ws + OFF_WE1T);
  h16* we2pt = (h16*)(ws + OFF_WE2PT);

  const size_t NX = (size_t)BB * NN * HIDD;   // 2097152
  const size_t NW = 512 * 512;                // 262144
  const size_t TOTAL = NX + 4 * NW + 64 * 32 + 16 * 64;
  size_t stride = (size_t)gridDim.x * blockDim.x;
  for (size_t t = (size_t)blockIdx.x * blockDim.x + threadIdx.x; t < TOTAL; t += stride) {
    if (t < NX) { xh[t] = (h16)x[t]; continue; }
    size_t u = t - NX;
    if (u < 4 * NW) {                          // transpose: wt[n*512+k] = W[k*512+n]
      int w = (int)(u / NW);
      size_t v = u - (size_t)w * NW;
      size_t n = v >> 9, k = v & 511;
      wt[w][v] = (h16)W[w][k * 512 + n];
      continue;
    }
    u -= 4 * NW;
    if (u < 64 * 32) {                         // we1t[n*32+k] = We1[k*64+n]
      size_t n = u >> 5, k = u & 31;
      we1t[u] = (h16)We1[k * 64 + n];
      continue;
    }
    u -= 64 * 32;
    {                                          // we2pt[n*64+k] = n<8 ? We2[k*8+n] : 0
      size_t n = u >> 6, k = u & 63;
      we2pt[u] = (n < 8) ? (h16)We2[k * 8 + n] : (h16)0.0f;
    }
  }
}

// ---------------- kernel 2: QKV projection GEMM (WMMA) ----------------
// grid = 256 row-tiles; 256 threads (8 waves). Writes Q,K f16 (B,H,N,D); V^T f16 (B,H,D,N).
__global__ __launch_bounds__(256) void qkv_kernel(char* __restrict__ ws,
                                                  const float* __restrict__ bq,
                                                  const float* __restrict__ bk,
                                                  const float* __restrict__ bv,
                                                  const int* __restrict__ ct,
                                                  const float* __restrict__ ct_key_emb) {
  const h16* xh  = (const h16*)(ws + OFF_XH);
  const h16* Wqt = (const h16*)(ws + OFF_WQT);
  const h16* Wkt = (const h16*)(ws + OFF_WKT);
  const h16* Wvt = (const h16*)(ws + OFF_WVT);
  h16* Qh = (h16*)(ws + OFF_QH);
  h16* Kh = (h16*)(ws + OFF_KH);
  h16* Vt = (h16*)(ws + OFF_VT);

  __shared__ h16 xt[16 * 512];
  int r0 = blockIdx.x * 16;
  int b = r0 >> 9, ns0 = r0 & 511;
  for (int t = threadIdx.x; t < 1024; t += 256)          // vector-stage the x tile
    ((v8h*)xt)[t] = ((const v8h*)(xh + (size_t)r0 * 512))[t];
  __syncthreads();

  int wave = threadIdx.x >> 5, lane = threadIdx.x & 31;
  int hi = lane >> 4, l16 = lane & 15;
  int cti = ct[b];

  for (int t = wave; t < 96; t += 8) {                   // 3 matrices x 32 col-tiles
    int mat = t >> 5, n0 = (t & 31) * 16;
    const h16* Wt = (mat == 0) ? Wqt : (mat == 1 ? Wkt : Wvt);
    const float* bias = (mat == 0) ? bq : (mat == 1 ? bk : bv);
    int C = n0 + l16;
    v8f acc = {};
    for (int ks = 0; ks < 16; ++ks) {
      int k0 = ks * 32;
      v16h a  = ldfrag(xt + l16 * 512 + k0, hi);
      v16h bf = ldfrag(Wt + (size_t)C * 512 + k0, hi);
      acc = wmma16(a, bf, acc);
    }
    float bC = bias[C];
    if (mat == 1) bC += ct_key_emb[(size_t)cti * HIDD + C];
    int hh = C >> 6, d = C & 63;
    V8F o; o.v = acc;
#pragma unroll
    for (int r = 0; r < 8; ++r) {
      int m = r + 8 * hi;
      float val = o.f[r] + bC;
      if (mat == 0)
        Qh[(((size_t)(b * HH + hh)) * NN + ns0 + m) * DD + d] = (h16)val;
      else if (mat == 1)
        Kh[(((size_t)(b * HH + hh)) * NN + ns0 + m) * DD + d] = (h16)val;
      else
        Vt[(((size_t)(b * HH + hh)) * DD + d) * NN + ns0 + m] = (h16)val;
    }
  }
}

// ---------------- kernel 3: edge-feature MLP bias (WMMA), eb f16 (B,N,N,H) ----------------
__global__ __launch_bounds__(256) void edge_kernel(const float* __restrict__ ef,
                                                   const float* __restrict__ be1,
                                                   const float* __restrict__ be2,
                                                   char* __restrict__ ws) {
  const h16* we1t  = (const h16*)(ws + OFF_WE1T);   // 64 x 32 (n-major)
  const h16* we2pt = (const h16*)(ws + OFF_WE2PT);  // 16 x 64 (n-major, rows>=8 zero)
  h16* ebh = (h16*)(ws + OFF_EBH);

  __shared__ h16 efl[128 * 32];                     // 8 waves x 16 rows, f16
  __shared__ h16 hid[8][16 * 64];

  int wave = threadIdx.x >> 5, lane = threadIdx.x & 31;
  int hi = lane >> 4, l16 = lane & 15;
  size_t rowBase = (size_t)blockIdx.x * 128;        // 128 rows per block

  for (int t = threadIdx.x; t < 128 * 32; t += 256)
    efl[t] = (h16)ef[rowBase * EE + t];
  __syncthreads();

  // layer 1: K = 32, A-fragment reused across all 4 N-tiles
  v16h a1 = ldfrag(&efl[(wave * 16 + l16) * 32], hi);
#pragma unroll
  for (int nt = 0; nt < 4; ++nt) {
    int n0 = nt * 16;
    v16h bf = ldfrag(we1t + (n0 + l16) * 32, hi);
    v8f acc = {};
    acc = wmma16(a1, bf, acc);
    V8F o; o.v = acc;
#pragma unroll
    for (int r = 0; r < 8; ++r) {
      int m = r + 8 * hi;
      float xv = o.f[r] + be1[n0 + l16];
      xv = xv > 0.0f ? xv : (__expf(xv) - 1.0f);    // ELU
      hid[wave][m * 64 + n0 + l16] = (h16)xv;
    }
  }

  // layer 2: K = 64 -> 2 WMMAs (per-wave LDS, same-wave ordering is in-order)
  v8f acc2 = {};
#pragma unroll
  for (int ks = 0; ks < 2; ++ks) {
    int k0 = ks * 32;
    v16h a  = ldfrag(&hid[wave][l16 * 64 + k0], hi);
    v16h bf = ldfrag(we2pt + l16 * 64 + k0, hi);
    acc2 = wmma16(a, bf, acc2);
  }
  V8F o2; o2.v = acc2;
  size_t row0 = rowBase + (size_t)wave * 16;
  if (l16 < 8) {
#pragma unroll
    for (int r = 0; r < 8; ++r) {
      int m = r + 8 * hi;
      ebh[(row0 + (size_t)m) * HH + l16] = (h16)(o2.f[r] + be2[l16]);
    }
  }
}

// ---------------- kernel 4: attention; TDM-staged f16 K / V^T in LDS ----------------
// grid = B*32 = 256 workgroups; 256 threads (8 waves).
// dynamic LDS: K f16 64K | V^T f16 64K | S f32 32K | P f16 16K | ctx f32 4K = 180 KiB
__global__ __launch_bounds__(256) void attn_kernel(const int* __restrict__ ct,
                                                   const float* __restrict__ ct_bias_emb,
                                                   const unsigned char* __restrict__ block_mask,
                                                   const unsigned char* __restrict__ pad_mask,
                                                   char* __restrict__ ws,
                                                   float* __restrict__ out_mean) {
  extern __shared__ char smem[];
  h16*   Klds = (h16*)smem;                          // [512][64]
  h16*   Vtld = (h16*)(smem + 65536);                // [64][512]
  float* Slds = (float*)(smem + 131072);             // [16][512]
  h16*   Plds = (h16*)(smem + 163840);               // [16][512]
  float* ctx  = (float*)(smem + 180224);             // [16][64]

  int b  = blockIdx.x >> 5;
  int it = blockIdx.x & 31;
  int i0 = it * 16;
  int wave = threadIdx.x >> 5, lane = threadIdx.x & 31;
  int hi = lane >> 4, l16 = lane & 15;

  const h16* Qh = (const h16*)(ws + OFF_QH);
  const h16* Kh = (const h16*)(ws + OFF_KH);
  const h16* Vt = (const h16*)(ws + OFF_VT);
  const h16* ebh = (const h16*)(ws + OFF_EBH);
  h16* pre = (h16*)(ws + OFF_PRE);

  const float NEG_INF = -__builtin_inff();
  int cti = ct[b];

  float mean_acc[32];
#pragma unroll
  for (int i = 0; i < 32; ++i) mean_acc[i] = 0.0f;

  for (int h = 0; h < HH; ++h) {
    __syncthreads();                                  // LDS reuse across heads
    for (int t = threadIdx.x; t < 16 * 64; t += 256) ctx[t] = 0.0f;
    if (wave == 0) {                                  // one wave drives the TDM
      unsigned long long kg = (unsigned long long)(const void*)(Kh + ((size_t)(b * HH + h)) * NN * DD);
      unsigned long long vg = (unsigned long long)(const void*)(Vt + ((size_t)(b * HH + h)) * DD * NN);
      tdm_load_2d(kg, (unsigned int)(unsigned long long)Klds, 2048u, 16u, 1u);
      tdm_load_2d(vg, (unsigned int)(unsigned long long)Vtld, 2048u, 16u, 1u);
      __builtin_amdgcn_s_wait_tensorcnt(0);
    }
    __syncthreads();

    float ctb = ct_bias_emb[(size_t)cti * HH + h];
    const h16* Qrow = Qh + ((size_t)(b * HH + h)) * NN * DD + (size_t)(i0 + l16) * DD;
    v16h aQ0 = ldfrag(Qrow, hi);
    v16h aQ1 = ldfrag(Qrow + 32, hi);

    // S = Q @ K^T  (16 x 512) with biases + masks
    for (int jt = wave; jt < 32; jt += 8) {
      int j0 = jt * 16;
      v8f acc = {};
      acc = wmma16(aQ0, ldfrag(&Klds[(j0 + l16) * DD], hi), acc);
      acc = wmma16(aQ1, ldfrag(&Klds[(j0 + l16) * DD + 32], hi), acc);
      V8F s8; s8.v = acc;
      int j = j0 + l16;
      bool padj = pad_mask[b * NN + j];
#pragma unroll
      for (int r = 0; r < 8; ++r) {
        int m = r + 8 * hi;
        int i = i0 + m;
        float s = s8.f[r] * 0.125f + ctb
                + (float)ebh[(((size_t)b * NN + i) * NN + j) * HH + h];
        bool msk = block_mask[((size_t)b * NN + i) * NN + j]
                || padj || pad_mask[b * NN + i];
        Slds[m * NN + j] = msk ? NEG_INF : s;
      }
    }
    __syncthreads();

    // softmax: wave -> rows {2w, 2w+1}; lane covers 16 columns each
#pragma unroll
    for (int rs = 0; rs < 2; ++rs) {
      int m = wave * 2 + rs;
      float vals[16];
      float mx = NEG_INF;
#pragma unroll
      for (int t = 0; t < 16; ++t) {
        vals[t] = Slds[m * NN + lane + 32 * t];
        mx = fmaxf(mx, vals[t]);
      }
#pragma unroll
      for (int off = 16; off >= 1; off >>= 1) mx = fmaxf(mx, __shfl_xor(mx, off, 32));
      bool dead = (mx < -1e37f);                      // fully-masked row -> zeros
      float sum = 0.0f;
#pragma unroll
      for (int t = 0; t < 16; ++t) {
        float p = dead ? 0.0f : __expf(vals[t] - mx);
        vals[t] = p; sum += p;
      }
#pragma unroll
      for (int off = 16; off >= 1; off >>= 1) sum += __shfl_xor(sum, off, 32);
      float inv = (sum > 0.0f) ? 1.0f / sum : 0.0f;
#pragma unroll
      for (int t = 0; t < 16; ++t) {
        float p = vals[t] * inv;
        Plds[m * NN + lane + 32 * t] = (h16)p;
        mean_acc[rs * 16 + t] += p * 0.125f;          // mean over H heads
      }
    }
    __syncthreads();

    // ctx = P @ V : wave -> (d-tile = w&3, k-half = w>>2); LDS f32 atomic reduce
    {
      int n0 = (wave & 3) * 16;
      int khalf = wave >> 2;
      v8f acc = {};
      for (int ks = 0; ks < 8; ++ks) {
        int k0 = (khalf * 8 + ks) * 32;
        v16h a  = ldfrag(&Plds[l16 * NN + k0], hi);
        v16h bf = ldfrag(&Vtld[(n0 + l16) * NN + k0], hi);
        acc = wmma16(a, bf, acc);
      }
      V8F o; o.v = acc;
#pragma unroll
      for (int r = 0; r < 8; ++r)
        atomicAdd(&ctx[(r + 8 * hi) * DD + n0 + l16], o.f[r]);
    }
    __syncthreads();

    for (int t = threadIdx.x; t < 16 * 64; t += 256) {
      int m = t >> 6, d = t & 63;
      pre[((size_t)b * NN + i0 + m) * HIDD + h * DD + d] = (h16)ctx[t];
    }
  }

  // attention mean output (each (b,i,j) owned by exactly one thread)
#pragma unroll
  for (int rs = 0; rs < 2; ++rs) {
    int i = i0 + wave * 2 + rs;
#pragma unroll
    for (int t = 0; t < 16; ++t)
      out_mean[((size_t)b * NN + i) * NN + lane + 32 * t] = mean_acc[rs * 16 + t];
  }
}

// ---------------- kernel 5: output projection (WMMA) ----------------
__global__ __launch_bounds__(256) void outproj_kernel(char* __restrict__ ws,
                                                      const float* __restrict__ bo,
                                                      float* __restrict__ out) {
  const h16* Wot = (const h16*)(ws + OFF_WOT);
  const h16* pre = (const h16*)(ws + OFF_PRE);

  __shared__ h16 xt[16 * 512];
  int r0 = blockIdx.x * 16;
  for (int t = threadIdx.x; t < 1024; t += 256)
    ((v8h*)xt)[t] = ((const v8h*)(pre + (size_t)r0 * 512))[t];
  __syncthreads();

  int wave = threadIdx.x >> 5, lane = threadIdx.x & 31;
  int hi = lane >> 4, l16 = lane & 15;

  for (int ctile = wave; ctile < 32; ctile += 8) {
    int n0 = ctile * 16;
    int C = n0 + l16;
    v8f acc = {};
    for (int ks = 0; ks < 16; ++ks) {
      int k0 = ks * 32;
      v16h a  = ldfrag(xt + l16 * 512 + k0, hi);
      v16h bf = ldfrag(Wot + (size_t)C * 512 + k0, hi);
      acc = wmma16(a, bf, acc);
    }
    V8F o; o.v = acc;
#pragma unroll
    for (int r = 0; r < 8; ++r)
      out[(size_t)(r0 + r + 8 * hi) * HIDD + C] = o.f[r] + bo[C];
  }
}

// ---------------- host-side launcher ----------------
extern "C" void kernel_launch(void* const* d_in, const int* in_sizes, int n_in,
                              void* d_out, int out_size, void* d_ws, size_t ws_size,
                              hipStream_t stream) {
  (void)in_sizes; (void)n_in; (void)out_size; (void)ws_size;
  const float* x             = (const float*)d_in[0];
  const int*   cancer_type   = (const int*)d_in[1];
  const float* edge_features = (const float*)d_in[2];
  const unsigned char* block_mask = (const unsigned char*)d_in[3];
  const unsigned char* pad_mask   = (const unsigned char*)d_in[4];
  const float* Wq = (const float*)d_in[5];   const float* bq = (const float*)d_in[6];
  const float* Wk = (const float*)d_in[7];   const float* bk = (const float*)d_in[8];
  const float* Wv = (const float*)d_in[9];   const float* bv = (const float*)d_in[10];
  const float* Wo = (const float*)d_in[11];  const float* bo = (const float*)d_in[12];
  const float* We1 = (const float*)d_in[13]; const float* be1 = (const float*)d_in[14];
  const float* We2 = (const float*)d_in[15]; const float* be2 = (const float*)d_in[16];
  const float* ct_key_emb  = (const float*)d_in[17];
  const float* ct_bias_emb = (const float*)d_in[18];

  float* out      = (float*)d_out;
  float* out_mean = out + (size_t)BB * NN * HIDD;
  char* ws = (char*)d_ws;

  prep_kernel<<<4096, 256, 0, stream>>>(x, Wq, Wk, Wv, Wo, We1, We2, ws);
  qkv_kernel<<<(BB * NN) / 16, 256, 0, stream>>>(ws, bq, bk, bv, cancer_type, ct_key_emb);
  edge_kernel<<<(int)(((size_t)BB * NN * NN) / 128), 256, 0, stream>>>(edge_features, be1, be2, ws);
  size_t smem = 184320;  // 180 KiB dynamic LDS (< 320 KiB per WGP)
  attn_kernel<<<BB * 32, 256, smem, stream>>>(cancer_type, ct_bias_emb, block_mask, pad_mask, ws, out_mean);
  outproj_kernel<<<(BB * NN) / 16, 256, 0, stream>>>(ws, bo, out);
}
